// Transformer_4698694222296
// MI455X (gfx1250) — matrix operand order repaired and checked by reading, not compiled, not run on hardware
//
#include <hip/hip_runtime.h>
#include <hip/hip_bf16.h>
#include <math.h>

// Problem dims (fixed by reference)
#define BB   4
#define TT   2048
#define DIMM 256
#define HH   8
#define DD   32
#define FFNN 1024

typedef __attribute__((ext_vector_type(16))) __bf16 v16bf;
typedef __attribute__((ext_vector_type(8)))  __bf16 v8bf;
typedef __attribute__((ext_vector_type(8)))  float  v8f;

__device__ __forceinline__ v8f wmma_bf16(v16bf a, v16bf b, v8f c) {
  // D = A(16x32 bf16) * B(32x16 bf16) + C(16x16 f32)
  return __builtin_amdgcn_wmma_f32_16x16x32_bf16(false, a, false, b, (short)0, c,
                                                 false, false);
}

// 16 contiguous bf16 -> one B-fragment lane register set
__device__ __forceinline__ v16bf load16(const __bf16* p) {
  v8bf t0 = *(const v8bf*)p;
  v8bf t1 = *(const v8bf*)(p + 8);
  v16bf r;
#pragma unroll
  for (int e = 0; e < 8; ++e) { r[e] = t0[e]; r[8 + e] = t1[e]; }
  return r;
}

// two 8-contiguous bf16 segments -> one A-fragment lane register set
__device__ __forceinline__ v16bf load8x2(const __bf16* p0, const __bf16* p1) {
  v8bf t0 = *(const v8bf*)p0;
  v8bf t1 = *(const v8bf*)p1;
  v16bf r;
#pragma unroll
  for (int e = 0; e < 8; ++e) { r[e] = t0[e]; r[8 + e] = t1[e]; }
  return r;
}

enum { MODE_QK = 0, MODE_VT = 1, MODE_F32 = 2, MODE_LRELU = 3 };

// C = A(MxK f32) * W(NxK f32)^T + bias, A/W converted to bf16 on load.
// One wave computes one 16x16 output tile; 4 waves / block share the A tile.
template <int MODE, int N, int K>
__global__ __launch_bounds__(128) void wmma_gemm(
    const float* __restrict__ A, const float* __restrict__ W,
    const float* __restrict__ bias, void* __restrict__ outp) {
  const int lane = threadIdx.x & 31;
  const int wave = threadIdx.x >> 5;
  const int tile = blockIdx.x * 4 + wave;
  const int ntn  = N >> 4;
  const int tm   = tile / ntn;
  const int tn   = tile - tm * ntn;
  const int half = lane >> 4;
  const int l16  = lane & 15;
  const int arow = tm * 16 + l16;   // A row held by this lane (A-frag)
  const int wrow = tn * 16 + l16;   // output column = W row (B-frag)

  v8f acc = {};
  const float* ap = A + (size_t)arow * K;
  const float* wp = W + (size_t)wrow * K;
  for (int k0 = 0; k0 < K; k0 += 32) {
    v16bf a, b;
    {
      const float* a0 = ap + k0 + 16 * half;       // K = 16*half .. +7
      const float* a1 = ap + k0 + 16 * half + 8;   // K = 16*half+8 .. +7
#pragma unroll
      for (int e = 0; e < 8; ++e) { a[e] = (__bf16)a0[e]; a[8 + e] = (__bf16)a1[e]; }
    }
    {
      const float* w0 = wp + k0 + 16 * half;       // K = 16*half .. +15
#pragma unroll
      for (int e = 0; e < 16; ++e) b[e] = (__bf16)w0[e];
    }
    acc = wmma_bf16(a, b, acc);
  }
  const float bv = bias[wrow];
#pragma unroll
  for (int r = 0; r < 8; ++r) {
    const int mg = tm * 16 + r + 8 * half;         // global row (token)
    float c = acc[r] + bv;
    if constexpr (MODE == MODE_LRELU) c = (c >= 0.f) ? c : 0.01f * c;
    if constexpr (MODE == MODE_F32 || MODE == MODE_LRELU) {
      ((float*)outp)[(size_t)mg * N + wrow] = c;
    } else {
      const int bidx = mg >> 11;                   // / TT
      const int t    = mg & (TT - 1);
      const int h    = wrow >> 5;                  // / DD
      const int d    = wrow & (DD - 1);
      if constexpr (MODE == MODE_QK)
        ((__bf16*)outp)[(((size_t)bidx * HH + h) * TT + t) * DD + d] = (__bf16)c;
      else  // MODE_VT: [B,H,D,T]
        ((__bf16*)outp)[(((size_t)bidx * HH + h) * DD + d) * TT + t] = (__bf16)c;
    }
  }
}

// Fused flash attention: one wave per (b, h, 16-query tile).
// wid = ((b*128 + qt)*8 + h) so the 8 h-variants of one (b,qt) tile are
// adjacent waves -> they stream the SAME pos/pad cachelines together
// (pos is [b][q][k][h]; one h alone uses 1/8 of each 128B line).
// 64 keys per iteration: 4 score WMMAs + one softmax pass + 4 ctx WMMAs.
__global__ __launch_bounds__(128) void attn_kernel(
    const __bf16* __restrict__ q, const __bf16* __restrict__ k,
    const __bf16* __restrict__ vt, const float* __restrict__ pos,
    const float* __restrict__ pad, float* __restrict__ ctx) {
  __shared__ __bf16 lds_p[4][16 * 64];
  const int lane = threadIdx.x & 31;
  const int wave = threadIdx.x >> 5;
  const int wid  = blockIdx.x * 4 + wave;          // 0 .. B*(T/16)*H - 1
  const int h    = wid & (HH - 1);
  const int qt   = (wid >> 3) & (TT / 16 - 1);
  const int b    = wid >> 10;                      // / (128*8)
  const int bh   = b * HH + h;
  const int half = lane >> 4;
  const int l16  = lane & 15;

  const __bf16* qbase = q  + (size_t)bh * TT * DD;  // [T,D]
  const __bf16* kbase = k  + (size_t)bh * TT * DD;  // [T,D]
  const __bf16* vbase = vt + (size_t)bh * DD * TT;  // [D,T]

  const int qrow = qt * 16 + l16;
  const v16bf qa = load8x2(qbase + (size_t)qrow * DD + 16 * half,
                           qbase + (size_t)qrow * DD + 16 * half + 8);

  float rowmax[8], rowsum[8];
#pragma unroll
  for (int r = 0; r < 8; ++r) { rowmax[r] = -3.0e38f; rowsum[r] = 0.f; }
  v8f c0 = {}, c1 = {};                             // ctx acc: d 0..15 / 16..31
  const float scale = 0.17677669529663688f;         // 1/sqrt(32)
  __bf16* lp = lds_p[wave];

  for (int k0 = 0; k0 < TT; k0 += 64) {
    // scores: four 16x16 tiles S = Q(16x32) * K^T(32x16)
    v8f sv[4];
#pragma unroll
    for (int sub = 0; sub < 4; ++sub) {
      v16bf kb = load16(kbase + (size_t)(k0 + sub * 16 + l16) * DD + 16 * half);
      v8f z = {};
      sv[sub] = wmma_bf16(qa, kb, z);
    }
    // bias, padding, scale, online softmax (C-frag row r -> q = qt*16+r+8*half)
#pragma unroll
    for (int r = 0; r < 8; ++r) {
      const size_t mg   = (size_t)qt * 16 + r + 8 * half;
      const size_t qoff = ((size_t)b * TT + mg) * TT;
      float xs[4];
#pragma unroll
      for (int sub = 0; sub < 4; ++sub) {
        const size_t kk = qoff + k0 + sub * 16 + l16;
        xs[sub] = (sv[sub][r] + pos[kk * HH + h]) * scale + pad[kk];
      }
      float tmax = fmaxf(fmaxf(xs[0], xs[1]), fmaxf(xs[2], xs[3]));
#pragma unroll
      for (int mm = 8; mm >= 1; mm >>= 1) tmax = fmaxf(tmax, __shfl_xor(tmax, mm, 32));
      const float nmax = fmaxf(rowmax[r], tmax);
      const float corr = __expf(rowmax[r] - nmax);
      rowmax[r] = nmax;
      float es[4], tsum = 0.f;
#pragma unroll
      for (int sub = 0; sub < 4; ++sub) { es[sub] = __expf(xs[sub] - nmax); tsum += es[sub]; }
#pragma unroll
      for (int mm = 8; mm >= 1; mm >>= 1) tsum += __shfl_xor(tsum, mm, 32);
      rowsum[r] = rowsum[r] * corr + tsum;
      c0[r] *= corr;
      c1[r] *= corr;
      // park P in LDS (row-major 16x64 bf16) for C-layout -> A-layout relayout
#pragma unroll
      for (int sub = 0; sub < 4; ++sub)
        lp[(r + 8 * half) * 64 + sub * 16 + l16] = (__bf16)es[sub];
    }
    asm volatile("s_wait_dscnt 0" ::: "memory");    // same-wave LDS RAW
    const v16bf pa0 = load8x2(lp + l16 * 64 + 16 * half,
                              lp + l16 * 64 + 8 + 16 * half);
    const v16bf pa1 = load8x2(lp + l16 * 64 + 32 + 16 * half,
                              lp + l16 * 64 + 40 + 16 * half);
    // ctx(16q x 32d) += P(16x64) * V(64x32): two key blocks x two d sub-tiles
    {
      v16bf vb;
      vb = load16(vbase + (size_t)l16 * TT + k0 + 16 * half);
      c0 = wmma_bf16(pa0, vb, c0);
      vb = load16(vbase + (size_t)(16 + l16) * TT + k0 + 16 * half);
      c1 = wmma_bf16(pa0, vb, c1);
      vb = load16(vbase + (size_t)l16 * TT + k0 + 32 + 16 * half);
      c0 = wmma_bf16(pa1, vb, c0);
      vb = load16(vbase + (size_t)(16 + l16) * TT + k0 + 32 + 16 * half);
      c1 = wmma_bf16(pa1, vb, c1);
    }
  }
#pragma unroll
  for (int r = 0; r < 8; ++r) {
    const size_t mg  = (size_t)qt * 16 + r + 8 * half;
    const float  inv = 1.f / rowsum[r];
    float* op = ctx + ((size_t)b * TT + mg) * DIMM + h * DD;
    op[l16]      = c0[r] * inv;
    op[16 + l16] = c1[r] * inv;
  }
}

// out = LayerNorm(xa + xb) * g + be   (one wave per token, 8 f32/lane)
__global__ __launch_bounds__(128) void ln_kernel(
    const float* __restrict__ xa, const float* __restrict__ xb,
    const float* __restrict__ g, const float* __restrict__ be,
    float* __restrict__ out) {
  const int lane  = threadIdx.x & 31;
  const int wave  = threadIdx.x >> 5;
  const int token = blockIdx.x * 4 + wave;
  const float* pa = xa + (size_t)token * DIMM;
  const float* pb = xb + (size_t)token * DIMM;
  float v[8];
  float s = 0.f;
#pragma unroll
  for (int i = 0; i < 8; ++i) { v[i] = pa[lane + 32 * i] + pb[lane + 32 * i]; s += v[i]; }
#pragma unroll
  for (int mm = 16; mm >= 1; mm >>= 1) s += __shfl_xor(s, mm, 32);
  const float mu = s * (1.f / 256.f);
  float qv = 0.f;
#pragma unroll
  for (int i = 0; i < 8; ++i) { const float d = v[i] - mu; qv += d * d; }
#pragma unroll
  for (int mm = 16; mm >= 1; mm >>= 1) qv += __shfl_xor(qv, mm, 32);
  const float inv = rsqrtf(qv * (1.f / 256.f) + 1e-5f);
  float* op = out + (size_t)token * DIMM;
#pragma unroll
  for (int i = 0; i < 8; ++i)
    op[lane + 32 * i] = (v[i] - mu) * inv * g[lane + 32 * i] + be[lane + 32 * i];
}

extern "C" void kernel_launch(void* const* d_in, const int* in_sizes, int n_in,
                              void* d_out, int out_size, void* d_ws, size_t ws_size,
                              hipStream_t stream) {
  const float* x   = (const float*)d_in[0];
  const float* pos = (const float*)d_in[1];
  const float* pad = (const float*)d_in[2];
  const float* Wq  = (const float*)d_in[3];
  const float* bq  = (const float*)d_in[4];
  const float* Wk  = (const float*)d_in[5];
  const float* bk  = (const float*)d_in[6];
  const float* Wv  = (const float*)d_in[7];
  const float* bv  = (const float*)d_in[8];
  const float* Wo  = (const float*)d_in[9];
  const float* bo  = (const float*)d_in[10];
  const float* W1  = (const float*)d_in[11];
  const float* b1  = (const float*)d_in[12];
  const float* W2  = (const float*)d_in[13];
  const float* b2  = (const float*)d_in[14];
  const float* g1  = (const float*)d_in[15];
  const float* be1 = (const float*)d_in[16];
  const float* g2  = (const float*)d_in[17];
  const float* be2 = (const float*)d_in[18];

  char* ws = (char*)d_ws;
  __bf16* qbf = (__bf16*)(ws + (0ull  << 20));   //  4 MB  [B,H,T,D] bf16
  __bf16* kbf = (__bf16*)(ws + (4ull  << 20));   //  4 MB  [B,H,T,D] bf16
  __bf16* vtb = (__bf16*)(ws + (8ull  << 20));   //  4 MB  [B,H,D,T] bf16
  float*  ctx = (float*)(ws + (12ull << 20));    //  8 MB  [B,T,256] f32
  float*  tm1 = (float*)(ws + (20ull << 20));    //  8 MB  attn out
  float*  x1  = (float*)(ws + (28ull << 20));    //  8 MB  after LN1
  float*  hmd = (float*)(ws + (36ull << 20));    // 32 MB  FFN hidden
  float*  x2  = (float*)(ws + (68ull << 20));    //  8 MB  FFN out
  (void)in_sizes; (void)n_in; (void)out_size; (void)ws_size;

  const dim3 blk(128);
  // QKV projections: M=8192, N=256, K=256 -> 8192 tiles -> 2048 blocks
  wmma_gemm<MODE_QK, 256, 256><<<2048, blk, 0, stream>>>(x, Wq, bq, (void*)qbf);
  wmma_gemm<MODE_QK, 256, 256><<<2048, blk, 0, stream>>>(x, Wk, bk, (void*)kbf);
  wmma_gemm<MODE_VT, 256, 256><<<2048, blk, 0, stream>>>(x, Wv, bv, (void*)vtb);
  // flash attention: B*(T/16)*H = 4096 waves -> 1024 blocks
  attn_kernel<<<1024, blk, 0, stream>>>(qbf, kbf, vtb, pos, pad, ctx);
  // output projection: M=8192, N=256, K=256
  wmma_gemm<MODE_F32, 256, 256><<<2048, blk, 0, stream>>>(ctx, Wo, bo, (void*)tm1);
  // x1 = LN(x + attn_out)
  ln_kernel<<<2048, blk, 0, stream>>>(x, tm1, g1, be1, x1);
  // FFN1 + LeakyReLU: M=8192, N=1024, K=256 -> 32768 tiles -> 8192 blocks
  wmma_gemm<MODE_LRELU, 1024, 256><<<8192, blk, 0, stream>>>(x1, W1, b1, (void*)hmd);
  // FFN2: M=8192, N=256, K=1024
  wmma_gemm<MODE_F32, 256, 1024><<<2048, blk, 0, stream>>>(hmd, W2, b2, (void*)x2);
  // out = LN(x1 + x2)
  ln_kernel<<<2048, blk, 0, stream>>>(x1, x2, g2, be2, (float*)d_out);
}
